// NonLoacalInteraction_67439576482312
// MI455X (gfx1250) — matrix operand order repaired and checked by reading, not compiled, not run on hardware
//
#include <hip/hip_runtime.h>

typedef __attribute__((ext_vector_type(16))) _Float16 v16h;
typedef __attribute__((ext_vector_type(8)))  float    v8f;

#define WMMA_F16(A, B, C) \
  __builtin_amdgcn_wmma_f32_16x16x32_f16(false, (A), false, (B), (short)0, (C), false, false)

static constexpr int F = 128;
static constexpr int N = 2048;
static constexpr int BATCH = 16;

// ---------------------------------------------------------------------------
// Kernel 1: QKV projection.  y = x @ W^T + b, emitted as f16.
//   - one wave per 16(row) x 64(outfeat) strip: A operand (x rows) is loaded
//     and converted to f16 ONCE, reused across 4 column tiles (16 WMMAs/wave).
//   - Q, K written row-major f16 (b,n,f); V written TRANSPOSED f16 (b,f,n)
//     so the PV matmul's B-operand is a contiguous per-lane load.
// waves = 3 proj * 2048 rowtiles * 2 colgroups; block = 256 (8 waves)
// ---------------------------------------------------------------------------
__global__ __launch_bounds__(256) void qkv_proj_kernel(
    const float* __restrict__ x,
    const float* __restrict__ Wq, const float* __restrict__ bq,
    const float* __restrict__ Wk, const float* __restrict__ bk,
    const float* __restrict__ Wv, const float* __restrict__ bv,
    _Float16* __restrict__ qout, _Float16* __restrict__ kout,
    _Float16* __restrict__ vtout)
{
  const int lane = threadIdx.x & 31;
  const int wave = threadIdx.x >> 5;
  const int wid  = blockIdx.x * 8 + wave;       // 0 .. 3*2048*2-1
  const int proj = wid / ((BATCH * N / 16) * 2);
  const int rem  = wid % ((BATCH * N / 16) * 2);
  const int g0   = (rem >> 1) * 16;             // global row (b*N+n)
  const int o0   = (rem & 1) * 64;              // output feature base of strip

  const float* W    = (proj == 0) ? Wq : (proj == 1) ? Wk : Wv;
  const float* bias = (proj == 0) ? bq : (proj == 1) ? bk : bv;

  const int mrow = lane & 15;
  const int hi   = lane >> 4;                   // lane half select

  // ---- A operand: 16 rows of x, converted once, reused for 4 col tiles ---
  const float* arow = x + (size_t)(g0 + mrow) * F;
  v16h Aop[4];
#pragma unroll
  for (int c = 0; c < 4; ++c) {
    const int ka = c * 32 + hi * 16;            // A: K = ka..ka+15
#pragma unroll
    for (int i = 0; i < 8; ++i) {
      Aop[c][i]     = (_Float16)arow[ka + i];
      Aop[c][i + 8] = (_Float16)arow[ka + 8 + i];
    }
  }

  // ---- 4 column tiles: B[K=f][N=o] = W[o][f] -----------------------------
  v8f acc[4];
#pragma unroll
  for (int ft = 0; ft < 4; ++ft) acc[ft] = (v8f){};
#pragma unroll
  for (int ft = 0; ft < 4; ++ft) {
    const float* brow = W + (size_t)(o0 + ft * 16 + mrow) * F;
#pragma unroll
    for (int c = 0; c < 4; ++c) {
      const int kb = c * 32 + hi * 16;          // B: K = kb..kb+15
      v16h b;
#pragma unroll
      for (int i = 0; i < 16; ++i) b[i] = (_Float16)brow[kb + i];
      acc[ft] = WMMA_F16(Aop[c], b, acc[ft]);
    }
  }

  // ---- bias + store ------------------------------------------------------
#pragma unroll
  for (int ft = 0; ft < 4; ++ft) {
    const int o = o0 + ft * 16 + mrow;          // lane's output column
    const float bval = bias[o];
    if (proj < 2) {
      _Float16* dst = (proj == 0) ? qout : kout;
#pragma unroll
      for (int r = 0; r < 8; ++r) {
        const int nloc = r + hi * 8;            // C-layout row for VGPR r
        dst[(size_t)(g0 + nloc) * F + o] = (_Float16)(acc[ft][r] + bval);
      }
    } else {
      const int b_  = g0 / N;
      const int n0_ = g0 % N;
#pragma unroll
      for (int r = 0; r < 8; ++r) {
        const int nloc = r + hi * 8;
        vtout[(size_t)b_ * F * N + (size_t)o * N + n0_ + nloc] =
            (_Float16)(acc[ft][r] + bval);
      }
    }
  }
}

// ---------------------------------------------------------------------------
// Kernel 2: flash attention. One wave owns a 16-query tile:
//   - Q resident as 4 A-operands (16x128 f16)
//   - loop keys in blocks of 32: 8 score WMMAs, online softmax (exp2 domain,
//     matching v_exp_f32's native base) via __shfl_xor(width=16) row
//     reductions, P staged f16 through per-wave LDS (store C-layout, reload
//     A-layout as 16B ds_load_b128), 8 PV WMMAs.
// block = 128 (4 waves), grid = (B*N/16)/4 = 512
// ---------------------------------------------------------------------------
__global__ __launch_bounds__(128) void flash_attn_kernel(
    const _Float16* __restrict__ q, const _Float16* __restrict__ kk,
    const _Float16* __restrict__ vt, float* __restrict__ out)
{
  __shared__ _Float16 plds[4 * 16 * 32];

  const int lane = threadIdx.x & 31;
  const int wave = threadIdx.x >> 5;
  const int qt   = blockIdx.x * 4 + wave;       // 0..2047 query tile
  const int b_   = qt >> 7;                     // 128 tiles per batch
  const int n0   = (qt & 127) * 16;
  const int mrow = lane & 15;
  const int hi   = lane >> 4;

  // softmax(x/T) == 2^(x*sc2) / sum 2^(x*sc2); v_exp_f32 is base-2 native
  const float sc2 = 1.4426950408889634f / 16.0f;   // log2(e)/sqrt(2*128)

  // ---- load Q tile into 4 A-operands -------------------------------------
  const _Float16* qrow = q + ((size_t)b_ * N + n0 + mrow) * F;
  v16h Qa[4];
#pragma unroll
  for (int c = 0; c < 4; ++c) {
    const int ka = c * 32 + hi * 16;
#pragma unroll
    for (int i = 0; i < 8; ++i) {
      Qa[c][i]     = qrow[ka + i];
      Qa[c][i + 8] = qrow[ka + 8 + i];
    }
  }

  v8f oacc[8];
#pragma unroll
  for (int ft = 0; ft < 8; ++ft) oacc[ft] = (v8f){};
  float m_run[8], l_run[8];
#pragma unroll
  for (int r = 0; r < 8; ++r) { m_run[r] = -3.0e38f; l_run[r] = 0.0f; }

  const _Float16* kbase = kk + (size_t)b_ * N * F;
  const _Float16* vbase = vt + (size_t)b_ * F * N;
  _Float16* pl = plds + wave * (16 * 32);       // private 16x32 f16 tile

  for (int m0 = 0; m0 < N; m0 += 32) {
    // ---- scores: two 16x16 tiles over the 32-key block -------------------
    v8f s0 = {}, s1 = {};
#pragma unroll
    for (int c = 0; c < 4; ++c) {
      const int kb = c * 32 + hi * 16;
      v16h bm0, bm1;
      const _Float16* kr0 = kbase + (size_t)(m0 + mrow) * F + kb;
      const _Float16* kr1 = kbase + (size_t)(m0 + 16 + mrow) * F + kb;
#pragma unroll
      for (int i = 0; i < 16; ++i) { bm0[i] = kr0[i]; bm1[i] = kr1[i]; }
      s0 = WMMA_F16(Qa[c], bm0, s0);
      s1 = WMMA_F16(Qa[c], bm1, s1);
    }

    // ---- online softmax update (base-2 domain) ---------------------------
    float fac[8], mnew[8], p0[8], p1[8];
#pragma unroll
    for (int r = 0; r < 8; ++r) {
      float v = fmaxf(s0[r], s1[r]);
      v = fmaxf(v, __shfl_xor(v, 1, 16));
      v = fmaxf(v, __shfl_xor(v, 2, 16));
      v = fmaxf(v, __shfl_xor(v, 4, 16));
      v = fmaxf(v, __shfl_xor(v, 8, 16));       // row max over 32 key cols
      mnew[r]  = fmaxf(m_run[r], v * sc2);
      fac[r]   = exp2f(m_run[r] - mnew[r]);
      m_run[r] = mnew[r];
    }
#pragma unroll
    for (int r = 0; r < 8; ++r) {
      p0[r] = exp2f(fmaf(s0[r], sc2, -mnew[r]));
      p1[r] = exp2f(fmaf(s1[r], sc2, -mnew[r]));
      float ps = p0[r] + p1[r];
      ps += __shfl_xor(ps, 1, 16);
      ps += __shfl_xor(ps, 2, 16);
      ps += __shfl_xor(ps, 4, 16);
      ps += __shfl_xor(ps, 8, 16);              // row sum over 32 key cols
      l_run[r] = fmaf(l_run[r], fac[r], ps);
    }
#pragma unroll
    for (int ft = 0; ft < 8; ++ft)
#pragma unroll
      for (int r = 0; r < 8; ++r) oacc[ft][r] *= fac[r];

    // ---- stage P (16x32 f16) via LDS: C-layout -> A-layout ---------------
#pragma unroll
    for (int r = 0; r < 8; ++r) {
      const int nloc = r + hi * 8;
      pl[nloc * 32 + mrow]      = (_Float16)p0[r];
      pl[nloc * 32 + 16 + mrow] = (_Float16)p1[r];
    }
    asm volatile("s_wait_dscnt 0" ::: "memory");

    v16h Pa;
#pragma unroll
    for (int i = 0; i < 8; ++i) {
      Pa[i]     = pl[mrow * 32 + hi * 16 + i];       // 16B contiguous
      Pa[i + 8] = pl[mrow * 32 + hi * 16 + 8 + i];   // 16B contiguous
    }

    // ---- PV: 8 WMMAs, B from transposed V (contiguous K=m per lane) ------
#pragma unroll
    for (int ft = 0; ft < 8; ++ft) {
      const _Float16* vr = vbase + (size_t)(ft * 16 + mrow) * N + m0 + hi * 16;
      v16h bm;
#pragma unroll
      for (int i = 0; i < 16; ++i) bm[i] = vr[i];
      oacc[ft] = WMMA_F16(Pa, bm, oacc[ft]);
    }
  }

  // ---- normalize and write fp32 output ----------------------------------
  float invl[8];
#pragma unroll
  for (int r = 0; r < 8; ++r) invl[r] = 1.0f / l_run[r];
#pragma unroll
  for (int ft = 0; ft < 8; ++ft) {
#pragma unroll
    for (int r = 0; r < 8; ++r) {
      const int nloc = r + hi * 8;
      out[((size_t)b_ * N + n0 + nloc) * F + ft * 16 + mrow] =
          oacc[ft][r] * invl[r];
    }
  }
}

// ---------------------------------------------------------------------------
extern "C" void kernel_launch(void* const* d_in, const int* in_sizes, int n_in,
                              void* d_out, int out_size, void* d_ws,
                              size_t ws_size, hipStream_t stream) {
  (void)in_sizes; (void)n_in; (void)out_size; (void)ws_size;
  const float* x  = (const float*)d_in[0];
  const float* Wq = (const float*)d_in[1];
  const float* bq = (const float*)d_in[2];
  const float* Wk = (const float*)d_in[3];
  const float* bk = (const float*)d_in[4];
  const float* Wv = (const float*)d_in[5];
  const float* bv = (const float*)d_in[6];
  float* out = (float*)d_out;

  const size_t elems = (size_t)BATCH * N * F;     // 4,194,304
  _Float16* qws  = (_Float16*)d_ws;               // 8 MB
  _Float16* kws  = qws + elems;                   // 8 MB
  _Float16* vtws = kws + elems;                   // 8 MB (transposed V)

  // 3 proj * 2048 rowtiles * 2 colgroups waves; 8 waves per block
  qkv_proj_kernel<<<3 * (BATCH * N / 16) * 2 / 8, 256, 0, stream>>>(
      x, Wq, bq, Wk, bk, Wv, bv, qws, kws, vtws);

  // 2048 query tiles, 4 waves per block
  flash_attn_kernel<<<(BATCH * N / 16) / 4, 128, 0, stream>>>(
      qws, kws, vtws, out);
}
